// MultiQueryAttention_45818711114007
// MI455X (gfx1250) — matrix operand order repaired and checked by reading, not compiled, not run on hardware
//
#include <hip/hip_runtime.h>

// ---------------------------------------------------------------------------
// Multi-Query Attention forward for MI455X (gfx1250, wave32, WMMA + TDM).
// Pipeline: f32->bf16 cast; q = x@Wq+bq; kv = x@Wkv+bkv; flash attention per
// head (online softmax, f32 accum); out = attn_out @ Wo + bo  (f32 result).
// Matmuls use v_wmma_f32_16x16x32_bf16; rectangular LDS staging uses the
// Tensor Data Mover (tensor_load_to_lds + s_wait_tensorcnt) where available.
// ---------------------------------------------------------------------------

typedef __attribute__((ext_vector_type(16))) __bf16       v16bf;
typedef __attribute__((ext_vector_type(8)))  float        v8f;
typedef __attribute__((ext_vector_type(4)))  unsigned int u32x4;
typedef __attribute__((ext_vector_type(4)))  float        f32x4;
typedef __attribute__((ext_vector_type(8)))  int          i32x8;
typedef __attribute__((ext_vector_type(4)))  int          i32x4;

union Frag {
  u32x4  u[2];
  v16bf  v;
  __bf16 e[16];
};

#define WMMA_BF16(a, b, c)                                                   \
  __builtin_amdgcn_wmma_f32_16x16x32_bf16(false, (a), false, (b), (short)0,  \
                                          (c), false, false)

// ---- Tensor Data Mover support (arity differs across toolchains) ----------
#if defined(__has_builtin)
#if __has_builtin(__builtin_amdgcn_tensor_load_to_lds) && \
    __has_builtin(__builtin_amdgcn_s_wait_tensorcnt)
#define USE_TDM 1
#endif
#endif
#ifndef USE_TDM
#define USE_TDM 0
#endif

#if __has_include(<hip/amd_detail/amd_gfx1250_TDM.h>)
#define TDM_6ARG 1   // therock-10.0 headers -> 6-arg builtin
#else
#define TDM_6ARG 0   // ROCm 7.2 -> 5-arg builtin
#endif

#if USE_TDM
// Issue one 2-D TDM tile load: global (tile_w x tile_h, elements of 2 bytes,
// row stride row_stride elements) -> LDS at byte offset lds_addr, with
// optional LDS row padding (pad_amount+1 DWORDs after every 2<<pad_interval
// DWORDs) to reproduce the bank-padded LDS layouts used by the WMMA readers.
__device__ __forceinline__ void tdm_load_tile_2d(
    unsigned lds_addr, const void* gptr, unsigned tile_w, unsigned tile_h,
    unsigned tensor_w, unsigned tensor_h, unsigned long long row_stride,
    unsigned pad_interval_code, unsigned pad_amount_code) {
  const unsigned long long ga = (unsigned long long)gptr;
  u32x4 g0;
  g0.x = 1u;                                          // count = 1 (valid D#)
  g0.y = lds_addr;                                    // LDS byte address
  g0.z = (unsigned)ga;                                // global_addr[31:0]
  g0.w = (unsigned)((ga >> 32) & 0x01FFFFFFull) | (2u << 30);  // type=2
  i32x8 g1;
  g1[0] = (int)((1u << 16) |                          // data_size = 2 bytes
                (1u << 20) |                          // pad_enable
                (pad_interval_code << 22) | (pad_amount_code << 25));
  g1[1] = (int)((tensor_w & 0xFFFFu) << 16);          // tensor_dim0[15:0]
  g1[2] = (int)((tensor_w >> 16) | ((tensor_h & 0xFFFFu) << 16));
  g1[3] = (int)((tensor_h >> 16) | (tile_w << 16));   // tile_dim0
  g1[4] = (int)(tile_h & 0xFFFFu);                    // tile_dim1 (tile_dim2=0)
  g1[5] = (int)(unsigned)(row_stride & 0xFFFFFFFFull);  // dim0 stride lo
  g1[6] = (int)(unsigned)((row_stride >> 32) & 0xFFFFull);
  g1[7] = 0;
  const i32x4 z4 = {0, 0, 0, 0};
#if TDM_6ARG
  const i32x8 z8 = {0, 0, 0, 0, 0, 0, 0, 0};
  __builtin_amdgcn_tensor_load_to_lds(g0, g1, z4, z4, z8, 0);
#else
  __builtin_amdgcn_tensor_load_to_lds(g0, g1, z4, z4, 0);
#endif
}
#endif  // USE_TDM

// ---------------------------- f32 -> bf16 cast -----------------------------
__global__ __launch_bounds__(256)
void cvt_bf16_kernel(const float* __restrict__ in, __bf16* __restrict__ out,
                     int n4) {
  int i = blockIdx.x * 256 + threadIdx.x;
  if (i >= n4) return;
  f32x4 f = ((const f32x4*)in)[i];
  out[4 * i + 0] = (__bf16)f.x;
  out[4 * i + 1] = (__bf16)f.y;
  out[4 * i + 2] = (__bf16)f.z;
  out[4 * i + 3] = (__bf16)f.w;
}

// ------------------------------- bf16 GEMM ---------------------------------
// C[M,N] = A[M,K] @ B[K,N] + bias.  128x128 tile / block (8 waves, each 32x64).
// LDS: A row-major [m][k] (staged by TDM with padding), B transposed [n][k].
template <bool OUT_BF16>
__global__ __launch_bounds__(256)
void gemm_bf16_kernel(const __bf16* __restrict__ A, const __bf16* __restrict__ B,
                      const float* __restrict__ bias, void* __restrict__ C,
                      int M, int N, int K) {
  // single LDS arena -> compile-time sub-offsets usable as TDM lds_addr
  __shared__ __attribute__((aligned(16))) unsigned char smem[20480];
  __bf16(*lA)[40]  = (__bf16(*)[40])(smem);          // 128 x 40  (10240 B @ 0)
  __bf16(*lBt)[40] = (__bf16(*)[40])(smem + 10240);  // 128 x 40  ([n][k])

  const int tid  = threadIdx.x;
  const int lane = tid & 31;
  const int wid  = tid >> 5;
  const int h    = lane >> 4;   // lane half selects K sub-range (ISA layout)
  const int l16  = lane & 15;
  const int wm   = wid & 3;     // 4 waves along M (32 rows each)
  const int wn   = wid >> 2;    // 2 waves along N (64 cols each)
  const int bm   = blockIdx.x << 7;
  const int bn   = blockIdx.y << 7;

  v8f acc[2][4];
  for (int mt = 0; mt < 2; ++mt)
    for (int nt = 0; nt < 4; ++nt)
      for (int r = 0; r < 8; ++r) acc[mt][nt][r] = 0.0f;

  const int arow = tid >> 1, ahalf = tid & 1;          // fallback A staging
  const int bkr  = tid >> 3, bnc   = (tid & 7) << 4;   // B tile: 32 x 128

  for (int k0 = 0; k0 < K; k0 += 32) {
#if USE_TDM
    if (wid == 0) {
      // A tile: 32 x 128 rows, row = 64B = 16 DW; LDS row 80B -> pad 4 DW
      // after every 16 DW (codes: interval 3 -> 16 DW, amount 3 -> 4 DW).
      tdm_load_tile_2d(/*lds*/0u, A + (size_t)bm * K + k0,
                       /*tile_w*/32u, /*tile_h*/128u,
                       /*tensor_w*/(unsigned)K, /*tensor_h*/(unsigned)M,
                       /*row_stride*/(unsigned long long)K,
                       /*pad_interval*/3u, /*pad_amount*/3u);
    }
#else
    {  // stage A (row-major) manually
      const u32x4* src = (const u32x4*)(A + (size_t)(bm + arow) * K + k0 + ahalf * 16);
      u32x4* dst = (u32x4*)&lA[arow][ahalf * 16];
      dst[0] = src[0];
      dst[1] = src[1];
    }
#endif
    {  // stage B transposed -> [n][k]
      union { u32x4 u[2]; __bf16 e[16]; } t;
      const u32x4* src = (const u32x4*)(B + (size_t)(k0 + bkr) * N + bn + bnc);
      t.u[0] = src[0];
      t.u[1] = src[1];
#pragma unroll
      for (int j = 0; j < 16; ++j) lBt[bnc + j][bkr] = t.e[j];
    }
    if (k0 + 32 < K)  // gfx1250 global_prefetch_b8 for next B k-tile
      __builtin_prefetch(B + (size_t)(k0 + 32 + bkr) * N + bn + bnc, 0, 0);
#if USE_TDM
    if (wid == 0) __builtin_amdgcn_s_wait_tensorcnt(0);
#endif
    __syncthreads();

    Frag af[2], bfr[4];
#pragma unroll
    for (int mt = 0; mt < 2; ++mt) {
      const int row = wm * 32 + mt * 16 + l16;           // lane = M row
      af[mt].u[0] = *(const u32x4*)&lA[row][h * 8];      // K = h*8 + 0..7
      af[mt].u[1] = *(const u32x4*)&lA[row][16 + h * 8]; // K = 16 + h*8 + 0..7
    }
#pragma unroll
    for (int nt = 0; nt < 4; ++nt) {
      const int col = wn * 64 + nt * 16 + l16;           // lane = N col
      bfr[nt].u[0] = *(const u32x4*)&lBt[col][h * 8];    // K = h*8 + 0..7
      bfr[nt].u[1] = *(const u32x4*)&lBt[col][16 + h * 8];
    }
#pragma unroll
    for (int mt = 0; mt < 2; ++mt)
#pragma unroll
      for (int nt = 0; nt < 4; ++nt)
        acc[mt][nt] = WMMA_BF16(af[mt].v, bfr[nt].v, acc[mt][nt]);
    __syncthreads();
  }

  // epilogue: C/D layout -> lane = N, VGPR r holds M row (r + 8*half)
#pragma unroll
  for (int mt = 0; mt < 2; ++mt)
#pragma unroll
    for (int nt = 0; nt < 4; ++nt) {
      const int n = bn + wn * 64 + nt * 16 + l16;
      const float bv = bias[n];
#pragma unroll
      for (int r = 0; r < 8; ++r) {
        const int m = bm + wm * 32 + mt * 16 + r + 8 * h;
        const float v = acc[mt][nt][r] + bv;
        if (OUT_BF16)
          ((__bf16*)C)[(size_t)m * N + n] = (__bf16)v;
        else
          ((float*)C)[(size_t)m * N + n] = v;
      }
    }
}

// -------------------------- flash MQA attention ----------------------------
// Block: 128 query rows of one (batch, head). 8 waves; wave owns 16 q-rows
// with its Q fragments resident in registers. KV chunked by 64 keys.
__global__ __launch_bounds__(256)
void mqa_flash_kernel(const __bf16* __restrict__ q, const __bf16* __restrict__ kv,
                      __bf16* __restrict__ o) {
  const int mtile = blockIdx.x;  // 16 q-row tiles
  const int head  = blockIdx.y;  // 16 heads
  const int b     = blockIdx.z;  // batch

  const int tid  = threadIdx.x;
  const int lane = tid & 31;
  const int wid  = tid >> 5;
  const int h    = lane >> 4;
  const int l16  = lane & 15;

  __shared__ __attribute__((aligned(16))) unsigned char smem[54272];
  __bf16(*lK)[136]     = (__bf16(*)[136])(smem);              // [key][d], 17408 B @ 0
  __bf16(*lVt)[72]     = (__bf16(*)[72])(smem + 17408);       // [d][key], 18432 B
  __bf16(*lP)[16][72]  = (__bf16(*)[16][72])(smem + 35840);   // per-wave probs

  const __bf16* kbase = kv + (size_t)(b * 2048) * 256;  // cols 0..127   = K
  const __bf16* vbase = kbase + 128;                    // cols 128..255 = V

  Frag qf[4];  // 16 rows x d=128 -> 4 k-step fragments, kept in VGPRs
  {
    const __bf16* qb =
        q + (size_t)(b * 2048 + mtile * 128 + wid * 16 + l16) * 2048 + head * 128;
#pragma unroll
    for (int ks = 0; ks < 4; ++ks) {
      qf[ks].u[0] = *(const u32x4*)(qb + ks * 32 + h * 8);
      qf[ks].u[1] = *(const u32x4*)(qb + ks * 32 + 16 + h * 8);
    }
  }

  float mrow[8], lrow[8];
  v8f oacc[8];
#pragma unroll
  for (int r = 0; r < 8; ++r) { mrow[r] = -1e30f; lrow[r] = 0.0f; }
#pragma unroll
  for (int dt = 0; dt < 8; ++dt)
    for (int r = 0; r < 8; ++r) oacc[dt][r] = 0.0f;

  const float scale = 0.08838834764831845f;  // 1/sqrt(128)
  const int jrow = tid >> 2, jpart = tid & 3;

  for (int nc = 0; nc < 2048; nc += 64) {
#if USE_TDM
    if (wid == 0) {
      // K chunk: 128 x 64 rows, row = 256B = 64 DW; LDS row 272B -> pad 4 DW
      // after every 64 DW (codes: interval 5 -> 64 DW, amount 3 -> 4 DW).
      tdm_load_tile_2d(/*lds*/0u, kbase + (size_t)nc * 256,
                       /*tile_w*/128u, /*tile_h*/64u,
                       /*tensor_w*/128u, /*tensor_h*/4096u,
                       /*row_stride*/256ull,
                       /*pad_interval*/5u, /*pad_amount*/3u);
    }
#else
    {  // stage K chunk [n'][d] manually
      const u32x4* src = (const u32x4*)(kbase + (size_t)(nc + jrow) * 256 + jpart * 32);
      u32x4* dst = (u32x4*)&lK[jrow][jpart * 32];
      dst[0] = src[0]; dst[1] = src[1]; dst[2] = src[2]; dst[3] = src[3];
    }
#endif
    {  // stage V chunk transposed [d][n']
      union { u32x4 u[4]; __bf16 e[32]; } t;
      const u32x4* src = (const u32x4*)(vbase + (size_t)(nc + jrow) * 256 + jpart * 32);
      t.u[0] = src[0]; t.u[1] = src[1]; t.u[2] = src[2]; t.u[3] = src[3];
#pragma unroll
      for (int e = 0; e < 32; ++e) lVt[jpart * 32 + e][jrow] = t.e[e];
    }
#if USE_TDM
    if (wid == 0) __builtin_amdgcn_s_wait_tensorcnt(0);
#endif
    __syncthreads();

    // S = q @ k^T : 16 rows x 64 keys per wave (4 N-tiles, 4 K-steps over d)
    v8f sacc[4];
#pragma unroll
    for (int nt = 0; nt < 4; ++nt) {
      for (int r = 0; r < 8; ++r) sacc[nt][r] = 0.0f;
#pragma unroll
      for (int ks = 0; ks < 4; ++ks) {
        Frag kf;
        kf.u[0] = *(const u32x4*)&lK[nt * 16 + l16][ks * 32 + h * 8];
        kf.u[1] = *(const u32x4*)&lK[nt * 16 + l16][ks * 32 + 16 + h * 8];
        sacc[nt] = WMMA_BF16(qf[ks].v, kf.v, sacc[nt]);
      }
    }

    // online softmax (row stats live across the 16-lane N-group)
    float mnew[8], corr[8];
#pragma unroll
    for (int r = 0; r < 8; ++r) {
      float mx = fmaxf(fmaxf(sacc[0][r], sacc[1][r]), fmaxf(sacc[2][r], sacc[3][r]));
#pragma unroll
      for (int off = 1; off < 16; off <<= 1) mx = fmaxf(mx, __shfl_xor(mx, off, 32));
      mx *= scale;
      const float mn = fmaxf(mrow[r], mx);
      corr[r] = __expf(mrow[r] - mn);
      mnew[r] = mn;
      mrow[r] = mn;
    }
#pragma unroll
    for (int nt = 0; nt < 4; ++nt)
#pragma unroll
      for (int r = 0; r < 8; ++r) {
        const float p = __expf(sacc[nt][r] * scale - mnew[r]);
        sacc[nt][r] = p;
        lP[wid][r + 8 * h][nt * 16 + l16] = (__bf16)p;  // accum-layout -> [m][n']
      }
#pragma unroll
    for (int r = 0; r < 8; ++r) {
      float s = sacc[0][r] + sacc[1][r] + sacc[2][r] + sacc[3][r];
#pragma unroll
      for (int off = 1; off < 16; off <<= 1) s += __shfl_xor(s, off, 32);
      lrow[r] = lrow[r] * corr[r] + s;
    }
#pragma unroll
    for (int dt = 0; dt < 8; ++dt)
#pragma unroll
      for (int r = 0; r < 8; ++r) oacc[dt][r] *= corr[r];

    // O += P @ V  (2 K-steps over the 64-key chunk, 8 d-tiles)
#pragma unroll
    for (int ks = 0; ks < 2; ++ks) {
      Frag pf;
      pf.u[0] = *(const u32x4*)&lP[wid][l16][ks * 32 + h * 8];
      pf.u[1] = *(const u32x4*)&lP[wid][l16][ks * 32 + 16 + h * 8];
#pragma unroll
      for (int dt = 0; dt < 8; ++dt) {
        Frag vf;
        vf.u[0] = *(const u32x4*)&lVt[dt * 16 + l16][ks * 32 + h * 8];
        vf.u[1] = *(const u32x4*)&lVt[dt * 16 + l16][ks * 32 + 16 + h * 8];
        oacc[dt] = WMMA_BF16(pf.v, vf.v, oacc[dt]);
      }
    }
    __syncthreads();
  }

  // epilogue: normalize and store in (B, N, H*dh) layout for the final GEMM
#pragma unroll
  for (int dt = 0; dt < 8; ++dt)
#pragma unroll
    for (int r = 0; r < 8; ++r) {
      const size_t m = (size_t)(b * 2048 + mtile * 128 + wid * 16 + r + 8 * h);
      o[m * 2048 + head * 128 + dt * 16 + l16] = (__bf16)(oacc[dt][r] / lrow[r]);
    }
}

// ------------------------------- launcher ----------------------------------
extern "C" void kernel_launch(void* const* d_in, const int* in_sizes, int n_in,
                              void* d_out, int out_size, void* d_ws, size_t ws_size,
                              hipStream_t stream) {
  const float* x   = (const float*)d_in[0];
  const float* Wq  = (const float*)d_in[1];
  const float* bq  = (const float*)d_in[2];
  const float* Wkv = (const float*)d_in[3];
  const float* bkv = (const float*)d_in[4];
  const float* Wo  = (const float*)d_in[5];
  const float* bo  = (const float*)d_in[6];
  (void)in_sizes; (void)n_in; (void)out_size; (void)ws_size;

  const int BN = 4096;  // B * N
  const int D  = 2048;
  const int KV = 256;   // 2 * HEAD_DIM

  __bf16* p = (__bf16*)d_ws;
  __bf16* x_bf   = p; p += (size_t)BN * D;
  __bf16* Wq_bf  = p; p += (size_t)D * D;
  __bf16* Wkv_bf = p; p += (size_t)D * KV;
  __bf16* Wo_bf  = p; p += (size_t)D * D;
  __bf16* q_bf   = p; p += (size_t)BN * D;
  __bf16* kv_bf  = p; p += (size_t)BN * KV;
  __bf16* o_bf   = p; p += (size_t)BN * D;   // ~70 MB total workspace

  auto cvt = [&](const float* src, __bf16* dst, size_t n) {
    int n4 = (int)(n / 4);
    cvt_bf16_kernel<<<(n4 + 255) / 256, 256, 0, stream>>>(src, dst, n4);
  };
  cvt(x,   x_bf,   (size_t)BN * D);
  cvt(Wq,  Wq_bf,  (size_t)D * D);
  cvt(Wkv, Wkv_bf, (size_t)D * KV);
  cvt(Wo,  Wo_bf,  (size_t)D * D);

  // q = x @ Wq + bq  (bf16 out)
  gemm_bf16_kernel<true><<<dim3(BN / 128, D / 128), 256, 0, stream>>>(
      x_bf, Wq_bf, bq, (void*)q_bf, BN, D, D);
  // kv = x @ Wkv + bkv (bf16 out; cols 0..127 = K, 128..255 = V)
  gemm_bf16_kernel<true><<<dim3(BN / 128, KV / 128), 256, 0, stream>>>(
      x_bf, Wkv_bf, bkv, (void*)kv_bf, BN, KV, D);
  // flash MQA
  mqa_flash_kernel<<<dim3(16, 16, 2), 256, 0, stream>>>(q_bf, kv_bf, o_bf);
  // out = attn_out @ Wo + bo (f32 out)
  gemm_bf16_kernel<false><<<dim3(BN / 128, D / 128), 256, 0, stream>>>(
      o_bf, Wo_bf, bo, d_out, BN, D, D);
}
